// dense_block_4449586118764
// MI455X (gfx1250) — compile-verified, matrix operand order checked by reading, not run on hardware
//
#include <hip/hip_runtime.h>

// ---------------------------------------------------------------------------
// Dense-block graph conv (4 layers) for MI455X / gfx1250.
// BN stats per produced block (fp32, deterministic 2-pass), BN+LeakyReLU fused
// into a bf16 activation buffer, one-ring conv via v_wmma_f32_16x16x32_bf16.
// Conv: one wave owns a 32-row x 64-col output tile (8 f32 accumulators),
// amortizing weight-fragment loads: 8 WMMA per 12 b128 loads per K-step.
// ---------------------------------------------------------------------------

typedef __attribute__((ext_vector_type(16))) __bf16 v16bf;
typedef __attribute__((ext_vector_type(8)))  float  v8f;

union Frag {
  v16bf v;
  uint4 u[2];
};

__device__ __forceinline__ unsigned short f2bf(float f) {
  unsigned int u = __float_as_uint(f);
  u += 0x7FFFu + ((u >> 16) & 1u);   // round-to-nearest-even
  return (unsigned short)(u >> 16);
}

// ---------------- per-block column statistics (deterministic) ---------------

// grid: STAT_BLOCKS x 256. part[b][0..63] = sum, part[b][64..127] = sumsq.
#define STAT_BLOCKS 256

__global__ void stats_partial_kernel(const float* __restrict__ x,
                                     float* __restrict__ part,
                                     long long n) {
  const int c  = threadIdx.x & 63;
  const int rg = threadIdx.x >> 6;          // 0..3 row groups
  float s = 0.f, s2 = 0.f;
  for (long long r = (long long)blockIdx.x * 4 + rg; r < n;
       r += (long long)gridDim.x * 4) {
    float v = x[r * 64 + c];
    s += v; s2 += v * v;
  }
  __shared__ float sh[512];
  sh[threadIdx.x]       = s;
  sh[256 + threadIdx.x] = s2;
  __syncthreads();
  if (threadIdx.x < 64) {
    float ts = 0.f, t2 = 0.f;
#pragma unroll
    for (int g = 0; g < 4; ++g) {
      ts += sh[g * 64 + c];
      t2 += sh[256 + g * 64 + c];
    }
    part[(size_t)blockIdx.x * 128 + c]      = ts;
    part[(size_t)blockIdx.x * 128 + 64 + c] = t2;
  }
}

// 1 block x 64 threads. stats[c] = mean, stats[64+c] = rsqrt(var + eps)
__global__ void stats_final_kernel(const float* __restrict__ part,
                                   float* __restrict__ stats,
                                   long long n) {
  const int c = threadIdx.x;
  float s = 0.f, s2 = 0.f;
  for (int p = 0; p < STAT_BLOCKS; ++p) {
    s  += part[(size_t)p * 128 + c];
    s2 += part[(size_t)p * 128 + 64 + c];
  }
  const float inv_n = 1.0f / (float)n;
  const float mu  = s * inv_n;
  const float var = s2 * inv_n - mu * mu;
  stats[c]      = mu;
  stats[64 + c] = rsqrtf(var + 1e-5f);
}

// ---------------- fused BN + LeakyReLU -> bf16 activations ------------------

__global__ void bn_act_kernel(const float* __restrict__ b0,
                              const float* __restrict__ b1,
                              const float* __restrict__ b2,
                              const float* __restrict__ b3,
                              const float* __restrict__ stats,   // [4][128]
                              const float* __restrict__ gamma,   // [cin]
                              const float* __restrict__ beta,    // [cin]
                              unsigned short* __restrict__ Y,    // [n, cin] bf16
                              long long n, int cin) {
  const float* src[4] = {b0, b1, b2, b3};
  const long long total  = n * (long long)cin;
  const long long stride = (long long)gridDim.x * blockDim.x;
  for (long long i = (long long)blockIdx.x * blockDim.x + threadIdx.x;
       i < total; i += stride) {
    long long row = i / cin;
    int cp  = (int)(i - row * cin);
    int blk = cp >> 6;
    int ch  = cp & 63;
    float val = src[blk][row * 64 + ch];
    float mu  = stats[blk * 128 + ch];
    float rs  = stats[blk * 128 + 64 + ch];
    float y   = (val - mu) * rs * gamma[cp] + beta[cp];
    y = (y > 0.f) ? y : 0.2f * y;
    Y[i] = f2bf(y);
  }
}

// ---------------- weight prep: fp32 [K,64] -> bf16 transposed [64,K] --------

__global__ void convert_w_kernel(const float* __restrict__ W,
                                 unsigned short* __restrict__ Wt,
                                 int ktot) {
  const int total = ktot * 64;
  for (int i = blockIdx.x * blockDim.x + threadIdx.x; i < total;
       i += gridDim.x * blockDim.x) {
    int col = i / ktot;
    int k   = i - col * ktot;
    Wt[i] = f2bf(W[(size_t)k * 64 + col]);
  }
}

// ---------------- one-ring conv via WMMA bf16 -------------------------------
// One wave = one 32-row tile x all 64 output cols (2x4 accumulators).
// A 32-wide K chunk lies inside one neighbor block (cin % 32 == 0), so each
// gathered A fragment is two contiguous b128 loads per lane.

template <int CIN>
__global__ __launch_bounds__(128) void conv_wmma_kernel(
    const unsigned short* __restrict__ Y,     // [n, CIN] bf16
    const int* __restrict__ neigh,            // [7n]
    const unsigned short* __restrict__ Wt,    // [64, 7*CIN] bf16 (W transposed)
    const float* __restrict__ bias,           // [64]
    float* __restrict__ out,                  // [n, 64]
    long long n) {
  constexpr int KTOT = 7 * CIN;
  const int lane = threadIdx.x & 31;
  const int wave = threadIdx.x >> 5;
  const long long tile = (long long)blockIdx.x * 4 + wave;  // 32-row tile
  const long long row0 = tile * 32;
  if (row0 >= n) return;                      // wave-uniform exit (EXEC all-1s)

  // 16-bit A layout: lanes 0-15 and 16-31 both hold rows 0..15 but different
  // K sub-chunks (lanes<16: K 0-7/16-23, lanes>=16: K 8-15/24-31).
  const int mrow   = lane & 15;
  const int kshift = (lane & 16) ? 8 : 0;
  const int ncol   = lane & 15;               // B/D column within 16-tile

  // Preload neighbor row offsets for both 16-row halves.
  long long nb[2][7];
#pragma unroll
  for (int h = 0; h < 2; ++h) {
    long long v = row0 + h * 16 + mrow;
    if (v >= n) v = n - 1;                    // clamp tail rows for loads
#pragma unroll
    for (int j = 0; j < 7; ++j)
      nb[h][j] = (long long)neigh[7 * v + j] * CIN;
  }

  v8f acc[2][4] = {};

  for (int k0 = 0; k0 < KTOT; k0 += 32) {
    const int j  = k0 / CIN;                  // neighbor index (chunk-uniform)
    const int c0 = k0 - j * CIN;              // channel offset in that block

    Frag a[2];
#pragma unroll
    for (int h = 0; h < 2; ++h) {
      const unsigned short* ap = Y + nb[h][j] + (c0 + kshift);
      a[h].u[0] = *(const uint4*)(ap);        // K = k0+ks .. +7
      a[h].u[1] = *(const uint4*)(ap + 16);   // K = k0+ks+16 .. +23
    }

    // Prefetch next K-chunk of this lane's weight column into cache.
    if (k0 + 32 < KTOT)
      __builtin_prefetch(Wt + (size_t)ncol * KTOT + (k0 + 32), 0, 1);

#pragma unroll
    for (int t = 0; t < 4; ++t) {
      const unsigned short* bp =
          Wt + (size_t)(t * 16 + ncol) * KTOT + (k0 + kshift);
      Frag b;
      b.u[0] = *(const uint4*)(bp);
      b.u[1] = *(const uint4*)(bp + 16);
      acc[0][t] = __builtin_amdgcn_wmma_f32_16x16x32_bf16(
          false, a[0].v, false, b.v, (short)0, acc[0][t], false, false);
      acc[1][t] = __builtin_amdgcn_wmma_f32_16x16x32_bf16(
          false, a[1].v, false, b.v, (short)0, acc[1][t], false, false);
    }
  }

  // 32-bit C/D layout: VGPR r -> M = r + (lane<16 ? 0 : 8), N = lane&15.
  const int rbase = (lane & 16) ? 8 : 0;
#pragma unroll
  for (int h = 0; h < 2; ++h) {
#pragma unroll
    for (int t = 0; t < 4; ++t) {
      const int col  = t * 16 + ncol;
      const float bv = bias[col];
#pragma unroll
      for (int r = 0; r < 8; ++r) {
        long long vr = row0 + h * 16 + rbase + r;
        if (vr < n) out[vr * 64 + col] = acc[h][t][r] + bv;
      }
    }
  }
}

// ---------------------------------------------------------------------------

static inline size_t align_up(size_t x) { return (x + 255) & ~(size_t)255; }

extern "C" void kernel_launch(void* const* d_in, const int* in_sizes, int n_in,
                              void* d_out, int out_size, void* d_ws, size_t ws_size,
                              hipStream_t stream) {
  (void)n_in; (void)out_size; (void)ws_size;

  const float* x     = (const float*)d_in[0];
  const int*   neigh = (const int*)d_in[1];
  // dict order: x, neigh, then (g,b,W,wb) for k=1..4
  const float* g[4]  = {(const float*)d_in[2],  (const float*)d_in[6],
                        (const float*)d_in[10], (const float*)d_in[14]};
  const float* bt[4] = {(const float*)d_in[3],  (const float*)d_in[7],
                        (const float*)d_in[11], (const float*)d_in[15]};
  const float* W[4]  = {(const float*)d_in[4],  (const float*)d_in[8],
                        (const float*)d_in[12], (const float*)d_in[16]};
  const float* wb[4] = {(const float*)d_in[5],  (const float*)d_in[9],
                        (const float*)d_in[13], (const float*)d_in[17]};

  const long long n = (long long)in_sizes[0] / 64;
  const int ktot[4] = {7 * 64, 7 * 128, 7 * 192, 7 * 256};

  // -------- workspace carve --------
  char* ws = (char*)d_ws;
  size_t off = 0;
  float* xbuf[3];                                // x1, x2, x3 (fp32)
  for (int s = 0; s < 3; ++s) {
    xbuf[s] = (float*)(ws + off);
    off = align_up(off + (size_t)n * 64 * sizeof(float));
  }
  unsigned short* Y = (unsigned short*)(ws + off);             // bf16 [n, 256] max
  off = align_up(off + (size_t)n * 256 * sizeof(unsigned short));
  unsigned short* Wt[4];
  for (int s = 0; s < 4; ++s) {
    Wt[s] = (unsigned short*)(ws + off);
    off = align_up(off + (size_t)ktot[s] * 64 * sizeof(unsigned short));
  }
  float* part = (float*)(ws + off);                            // [256][128]
  off = align_up(off + (size_t)STAT_BLOCKS * 128 * sizeof(float));
  float* stats = (float*)(ws + off);                           // [4][128]
  off = align_up(off + 4 * 128 * sizeof(float));

  // -------- weight conversion (bf16, transposed) --------
  for (int s = 0; s < 4; ++s)
    convert_w_kernel<<<256, 256, 0, stream>>>(W[s], Wt[s], ktot[s]);

  const float* blocks[4] = {x, xbuf[0], xbuf[1], xbuf[2]};
  float* outs[4] = {xbuf[0], xbuf[1], xbuf[2], (float*)d_out};

  const long long tiles = (n + 31) / 32;
  const int conv_blocks = (int)((tiles + 3) / 4);

  for (int s = 0; s < 4; ++s) {
    const int cin = 64 * (s + 1);
    // per-block column stats for the newest block (older blocks' stats reused)
    stats_partial_kernel<<<STAT_BLOCKS, 256, 0, stream>>>(blocks[s], part, n);
    stats_final_kernel<<<1, 64, 0, stream>>>(part, stats + s * 128, n);
    // fused BN + LeakyReLU -> bf16 activations Y [n, cin]
    bn_act_kernel<<<2048, 256, 0, stream>>>(blocks[0], blocks[1], blocks[2],
                                            blocks[3], stats, g[s], bt[s],
                                            Y, n, cin);
    // one-ring conv (gather + GEMM) via WMMA
    switch (s) {
      case 0: conv_wmma_kernel<64><<<conv_blocks, 128, 0, stream>>>(
                  Y, neigh, Wt[0], wb[0], outs[0], n); break;
      case 1: conv_wmma_kernel<128><<<conv_blocks, 128, 0, stream>>>(
                  Y, neigh, Wt[1], wb[1], outs[1], n); break;
      case 2: conv_wmma_kernel<192><<<conv_blocks, 128, 0, stream>>>(
                  Y, neigh, Wt[2], wb[2], outs[2], n); break;
      case 3: conv_wmma_kernel<256><<<conv_blocks, 128, 0, stream>>>(
                  Y, neigh, Wt[3], wb[3], outs[3], n); break;
    }
  }
}
